// InterpolationPredictionModel_91130616087187
// MI455X (gfx1250) — compile-verified, matrix operand order checked by reading, not run on hardware
//
#include <hip/hip_runtime.h>
#include <math.h>

// ---------------- problem constants ----------------
#define PB 8
#define PD 40
#define PT 192
#define PG 256
#define PU 256
#define PDEMO 16
#define F3D 120     // 3*D
#define NG 768      // 3*U
#define NTILES 48   // 768/16
#define MROWS 2048  // B*G
#define MTILES 128  // 2048/16

typedef __attribute__((ext_vector_type(16))) _Float16 v16h;
typedef __attribute__((ext_vector_type(8)))  _Float16 v8h;
typedef __attribute__((ext_vector_type(8)))  float    v8f;

// K index inside a 16x32 (or 32x16) f16 WMMA fragment:
// lane<16 half i: K = i (+8 if i>=8);  lane>=16: +8 more.  (ISA 7.12.2 tables)
__device__ __forceinline__ int frag_k(int i, bool hi) {
  return i + ((i >= 8) ? 8 : 0) + (hi ? 8 : 0);
}

// ---------------- weight -> B-fragment (f16) conversion ----------------
// dst layout: [ntile][ktile][lane][16 halves], contiguous 32B per lane.
__global__ void conv_bfrag_kernel(const float* __restrict__ W, _Float16* __restrict__ dst,
                                  int Ksrc, int KT) {
  int t = blockIdx.x * blockDim.x + threadIdx.x;
  int total = NTILES * KT * 32;
  if (t >= total) return;
  int lane = t & 31;
  int rest = t >> 5;
  int kt = rest % KT;
  int nt = rest / KT;
  bool hi = lane >= 16;
  int n = nt * 16 + (lane & 15);
  _Float16* d = dst + ((size_t)(nt * KT + kt) * 32 + lane) * 16;
#pragma unroll
  for (int i = 0; i < 16; ++i) {
    int k = kt * 32 + frag_k(i, hi);
    float v = (k < Ksrc) ? W[(size_t)k * NG + n] : 0.0f;
    d[i] = (_Float16)v;
  }
}

// ---------------- interpolation kernel ----------------
// one block per (b,d); thread g in [0,256); streaming logsumexp over T for both scales
__device__ __forceinline__ void lse_update(float l, float v, float& m, float& s, float& sy) {
  if (l > m) {
    float c = expf(m - l);   // exp(-inf)=0 on first hit
    s = s * c + 1.0f;
    sy = sy * c + v;
    m = l;
  } else {
    float e = expf(l - m);
    s += e;
    sy += e * v;
  }
}

__global__ __launch_bounds__(256) void interp_kernel(
    const float* __restrict__ times, const float* __restrict__ values,
    const float* __restrict__ meas, const float* __restrict__ grid,
    const float* __restrict__ ksic,
    float* __restrict__ y, float* __restrict__ w, float* __restrict__ ytr,
    float* __restrict__ mean) {
  __shared__ float ts[PT], vs[PT], ms[PT];
  __shared__ float red[256];
  int bd = blockIdx.x;
  int b = bd / PD, d = bd % PD;
  int tid = threadIdx.x;
  const size_t base = (size_t)bd * PT;
  for (int i = tid; i < PT; i += 256) {
    ts[i] = times[base + i];
    vs[i] = values[base + i];
    ms[i] = meas[base + i];
  }
  float alpha = log1pf(expf(ksic[d]));
  __syncthreads();

  int g = tid;
  float gv = grid[(size_t)b * PG + g];
  float m1 = -__builtin_inff(), s1 = 0.f, sy1 = 0.f;
  float m2 = -__builtin_inff(), s2 = 0.f, sy2 = 0.f;
  for (int t = 0; t < PT; ++t) {
    if (ms[t] <= 0.0f) continue;
    float dd = ts[t] - gv;
    float nrm = dd * dd;
    lse_update(-alpha * nrm, vs[t], m1, s1, sy1);
    lse_update(-10.0f * alpha * nrm, vs[t], m2, s2, sy2);
  }
  float yv = sy1 / s1;
  float wv = m1 + logf(s1);
  size_t o = (size_t)bd * PG + g;
  y[o] = yv;
  w[o] = wv;
  ytr[o] = sy2 / s2;

  // mean over G for this (b,d)
  red[tid] = yv;
  __syncthreads();
  for (int off = 128; off > 0; off >>= 1) {
    if (tid < off) red[tid] += red[tid + off];
    __syncthreads();
  }
  if (tid == 0) mean[bd] = red[0] * (1.0f / PG);
}

// ---------------- cross-channel kernel ----------------
// one block per (b,g); 64 threads; writes rnn_in row [120]
__global__ __launch_bounds__(64) void cross_kernel(
    const float* __restrict__ y, const float* __restrict__ w,
    const float* __restrict__ ytr, const float* __restrict__ mean,
    const float* __restrict__ crossW, float* __restrict__ rnn) {
  __shared__ float wv[PD], ym[PD], mn[PD];
  int blk = blockIdx.x;
  int b = blk >> 8, g = blk & 255;
  int tid = threadIdx.x;
  if (tid < PD) {
    size_t o = ((size_t)b * PD + tid) * PG + g;
    wv[tid] = w[o];
    mn[tid] = mean[b * PD + tid];
    ym[tid] = y[o] - mn[tid];
  }
  __syncthreads();
  if (tid < PD) {
    int e = tid;
    float mx = -__builtin_inff();
#pragma unroll 8
    for (int d = 0; d < PD; ++d) mx = fmaxf(mx, wv[d]);
    float se = 0.f;
#pragma unroll 8
    for (int d = 0; d < PD; ++d) se += expf(wv[d] - mx);
    float inv = 1.0f / se;
    float rep = 0.f;
    for (int d = 0; d < PD; ++d)
      rep += (expf(wv[d] - mx) * inv) * ym[d] * crossW[d * PD + e];
    rep += mn[e];
    size_t row = (size_t)b * PG + g;
    float* r = rnn + row * F3D;
    r[e] = rep;
    r[PD + e] = expf(wv[e]);
    r[2 * PD + e] = ytr[((size_t)b * PD + e) * PG + g] - rep;
  }
}

// ---------------- pack rnn_in -> A fragments (f16) ----------------
__global__ __launch_bounds__(256) void packA_kernel(const float* __restrict__ rnn,
                                                    _Float16* __restrict__ afrag) {
  int t = blockIdx.x * blockDim.x + threadIdx.x;   // 128 mtiles * 32 lanes
  int mt = t >> 5;
  int lane = t & 31;
  if (mt >= MTILES) return;
  int m = lane & 15;
  bool hi = lane >= 16;
  int row = mt * 16 + m;
#pragma unroll
  for (int kt = 0; kt < 4; ++kt) {
    _Float16* d = afrag + ((size_t)(mt * 4 + kt) * 32 + lane) * 16;
#pragma unroll
    for (int i = 0; i < 16; ++i) {
      int k = kt * 32 + frag_k(i, hi);
      float v = (k < F3D) ? rnn[(size_t)row * F3D + k] : 0.0f;
      d[i] = (_Float16)v;
    }
  }
}

// ---------------- x_proj GEMM: [2048,128pad] x [128pad,768] (WMMA) ----------------
__global__ __launch_bounds__(256) void xproj_kernel(const _Float16* __restrict__ afrag,
                                                    const _Float16* __restrict__ bxfrag,
                                                    const float* __restrict__ bi,
                                                    float* __restrict__ xproj) {
  int wid = blockIdx.x * 8 + (threadIdx.x >> 5);   // 6144 waves, one 16x16 tile each
  int lane = threadIdx.x & 31;
  int mt = wid / NTILES;
  int nt = wid % NTILES;
  v8f acc = {0.f, 0.f, 0.f, 0.f, 0.f, 0.f, 0.f, 0.f};
#pragma unroll
  for (int kt = 0; kt < 4; ++kt) {
    v16h a = *(const v16h*)(afrag + ((size_t)(mt * 4 + kt) * 32 + lane) * 16);
    v16h bf = *(const v16h*)(bxfrag + ((size_t)(nt * 4 + kt) * 32 + lane) * 16);
    acc = __builtin_amdgcn_wmma_f32_16x16x32_f16(false, a, false, bf, (short)0, acc,
                                                 false, false);
  }
  int col = nt * 16 + (lane & 15);
  float bias = bi[col];
  int rbase = mt * 16 + ((lane >= 16) ? 8 : 0);
#pragma unroll
  for (int j = 0; j < 8; ++j) {
    int row = rbase + j;
    xproj[(size_t)row * NG + col] = acc[j] + bias;
  }
}

// ---------------- demo encoder -> h0 ----------------
__global__ __launch_bounds__(256) void demo_kernel(const float* __restrict__ demo,
                                                   const float* __restrict__ W1,
                                                   const float* __restrict__ b1,
                                                   const float* __restrict__ W2,
                                                   const float* __restrict__ b2,
                                                   float* __restrict__ h0) {
  __shared__ float t1[PB * PU];
  int u = threadIdx.x;
#pragma unroll
  for (int b = 0; b < PB; ++b) {
    float s = b1[u];
    for (int k = 0; k < PDEMO; ++k) s += demo[b * PDEMO + k] * W1[k * PU + u];
    t1[b * PU + u] = fmaxf(s, 0.0f);
  }
  __syncthreads();
#pragma unroll
  for (int b = 0; b < PB; ++b) {
    float s = b2[u];
    for (int k = 0; k < PU; ++k) s += t1[b * PU + k] * W2[k * PU + u];
    h0[b * PU + u] = s;
  }
}

// ---------------- persistent masked-GRU kernel (WMMA recurrence) + output head ----------------
// h kept in f32 (gate math) with a packed f16 shadow (16 rows, rows 8..15 zero)
// so each WMMA A fragment is exactly two ds_load_b128 (contiguous 8-half runs).
__global__ __launch_bounds__(256) void gru_kernel(const float* __restrict__ xproj,
                                                  const _Float16* __restrict__ bhfrag,
                                                  const float* __restrict__ br,
                                                  const float* __restrict__ h0,
                                                  const int* __restrict__ lens,
                                                  const float* __restrict__ outW,
                                                  const float* __restrict__ outb,
                                                  float* __restrict__ out) {
  __shared__ float h_lds[PB * PU];                     // 8 KB (f32 state)
  __shared__ __align__(16) _Float16 h16_lds[16 * PU];  // 8 KB (f16 shadow, padded to 16 rows)
  __shared__ float hp_lds[PB * NG];                    // 24 KB
  __shared__ float br_s[NG];
  __shared__ int len_s[PB];
  __shared__ float part[256];

  int tid = threadIdx.x;
  int lane = tid & 31;
  int wave = tid >> 5;
  for (int i = tid; i < PB * PU; i += 256) {
    float v = h0[i];
    h_lds[i] = v;
    h16_lds[i] = (_Float16)v;
  }
  for (int i = PB * PU + tid; i < 16 * PU; i += 256) h16_lds[i] = (_Float16)0.0f;
  for (int i = tid; i < NG; i += 256) br_s[i] = br[i];
  if (tid < PB) len_s[tid] = lens[tid];
  __syncthreads();

  const int m = lane & 15;
  const bool hi = lane >= 16;
  const int ntbase = wave * 6;   // 48 N-tiles / 8 waves
  // A-fragment base for this lane: row m, K base (kt*32) + (hi ? 8 : 0)
  const _Float16* habase = h16_lds + m * PU + (hi ? 8 : 0);

  for (int g = 0; g < PG; ++g) {
    // ---- hp = h @ Wh  (16x16x32 WMMA, K=256 -> 8 k-steps, 6 N-tiles per wave) ----
    v8f acc[6];
    v8f zero = {0.f, 0.f, 0.f, 0.f, 0.f, 0.f, 0.f, 0.f};
#pragma unroll
    for (int t = 0; t < 6; ++t) acc[t] = zero;
#pragma unroll
    for (int kt = 0; kt < 8; ++kt) {
      // two contiguous 8-half runs: K = base+{0..7} and base+{16..23}
      const _Float16* hb = habase + kt * 32;
      union { v16h v; v8h h[2]; } ua;
      ua.h[0] = *(const v8h*)(hb);
      ua.h[1] = *(const v8h*)(hb + 16);
      v16h a = ua.v;
#pragma unroll
      for (int t = 0; t < 6; ++t) {
        v16h bf = *(const v16h*)(bhfrag +
                                 ((size_t)((ntbase + t) * 8 + kt) * 32 + lane) * 16);
        acc[t] = __builtin_amdgcn_wmma_f32_16x16x32_f16(false, a, false, bf, (short)0,
                                                        acc[t], false, false);
      }
    }
    if (lane < 16) {   // rows 0..7 (lanes>=16 carry padded rows 8..15)
#pragma unroll
      for (int t = 0; t < 6; ++t) {
        int col = (ntbase + t) * 16 + lane;
#pragma unroll
        for (int j = 0; j < 8; ++j) hp_lds[j * NG + col] = acc[t][j];
      }
    }
    __syncthreads();

    // ---- gates: 2048 (b,u) elements, 8 per thread ----
#pragma unroll
    for (int it = 0; it < 8; ++it) {
      int idx = tid + it * 256;
      int b = idx >> 8;
      int u = idx & 255;
      const float* xp = xproj + ((size_t)(b * PG + g)) * NG;
      float xz = xp[u], xr = xp[PU + u], xh = xp[2 * PU + u];
      float hz = hp_lds[b * NG + u] + br_s[u];
      float hr = hp_lds[b * NG + PU + u] + br_s[PU + u];
      float hh = hp_lds[b * NG + 2 * PU + u] + br_s[2 * PU + u];
      float z = 1.0f / (1.0f + expf(-(xz + hz)));
      float r = 1.0f / (1.0f + expf(-(xr + hr)));
      float cand = tanhf(xh + r * hh);
      float hold = h_lds[b * PU + u];
      float hnew = z * hold + (1.0f - z) * cand;
      float hsel = (g < len_s[b]) ? hnew : hold;
      h_lds[b * PU + u] = hsel;
      h16_lds[b * PU + u] = (_Float16)hsel;
    }
    __syncthreads();
  }

  // ---- output head: sigmoid(h @ out_W + out_b) ----
  {
    int b = tid >> 5, l = tid & 31;
    float s = 0.0f;
    for (int u = l; u < PU; u += 32) s += h_lds[b * PU + u] * outW[u];
    part[tid] = s;
    __syncthreads();
    if (tid < PB) {
      float t = 0.0f;
#pragma unroll
      for (int l2 = 0; l2 < 32; ++l2) t += part[tid * 32 + l2];
      out[tid] = 1.0f / (1.0f + expf(-(t + outb[0])));
    }
  }
}

// ---------------- host launcher ----------------
extern "C" void kernel_launch(void* const* d_in, const int* in_sizes, int n_in,
                              void* d_out, int out_size, void* d_ws, size_t ws_size,
                              hipStream_t stream) {
  (void)in_sizes; (void)n_in; (void)out_size; (void)ws_size;
  const float* demo   = (const float*)d_in[0];
  const float* times  = (const float*)d_in[1];
  const float* values = (const float*)d_in[2];
  const float* meas   = (const float*)d_in[3];
  const float* grid   = (const float*)d_in[4];
  const float* ksic   = (const float*)d_in[5];
  const float* crossW = (const float*)d_in[6];
  const float* dW1    = (const float*)d_in[7];
  const float* db1    = (const float*)d_in[8];
  const float* dW2    = (const float*)d_in[9];
  const float* db2    = (const float*)d_in[10];
  const float* gWx    = (const float*)d_in[11];
  const float* gWh    = (const float*)d_in[12];
  const float* gb     = (const float*)d_in[13];
  const float* outW   = (const float*)d_in[14];
  const float* outb   = (const float*)d_in[15];
  const int*   glen   = (const int*)d_in[16];
  float* out = (float*)d_out;

  char* p = (char*)d_ws;
  auto alloc = [&](size_t bytes) -> char* {
    char* r = p;
    p += (bytes + 255) & ~(size_t)255;
    return r;
  };
  float* y    = (float*)alloc((size_t)PB * PD * PG * 4);
  float* w    = (float*)alloc((size_t)PB * PD * PG * 4);
  float* ytr  = (float*)alloc((size_t)PB * PD * PG * 4);
  float* mean = (float*)alloc((size_t)PB * PD * 4);
  float* rnn  = (float*)alloc((size_t)MROWS * F3D * 4);
  float* xpj  = (float*)alloc((size_t)MROWS * NG * 4);
  float* h0   = (float*)alloc((size_t)PB * PU * 4);
  _Float16* afrag  = (_Float16*)alloc((size_t)MTILES * 4 * 32 * 16 * 2);
  _Float16* bxfrag = (_Float16*)alloc((size_t)NTILES * 4 * 32 * 16 * 2);
  _Float16* bhfrag = (_Float16*)alloc((size_t)NTILES * 8 * 32 * 16 * 2);

  // weight fragment conversion (L2-resident, done once per launch)
  conv_bfrag_kernel<<<(NTILES * 4 * 32 + 255) / 256, 256, 0, stream>>>(gWx, bxfrag, F3D, 4);
  conv_bfrag_kernel<<<(NTILES * 8 * 32 + 255) / 256, 256, 0, stream>>>(gWh, bhfrag, PU, 8);

  // interpolation + mean
  interp_kernel<<<PB * PD, 256, 0, stream>>>(times, values, meas, grid, ksic,
                                             y, w, ytr, mean);
  // cross-channel features -> rnn_in
  cross_kernel<<<PB * PG, 64, 0, stream>>>(y, w, ytr, mean, crossW, rnn);
  // pack rnn_in into WMMA A fragments
  packA_kernel<<<(MTILES * 32 + 255) / 256, 256, 0, stream>>>(rnn, afrag);
  // x_proj = rnn_in @ Wx + bi  (WMMA)
  xproj_kernel<<<(MTILES * NTILES) / 8, 256, 0, stream>>>(afrag, bxfrag, gb, xpj);
  // demo encoder -> h0
  demo_kernel<<<1, 256, 0, stream>>>(demo, dW1, db1, dW2, db2, h0);
  // persistent WMMA GRU + output head
  gru_kernel<<<1, 256, 0, stream>>>(xpj, bhfrag, gb + NG, h0, glen, outW, outb, out);
}
